// Dynamic_FC_257698038460
// MI455X (gfx1250) — matrix operand order repaired and checked by reading, not compiled, not run on hardware
//
#include <hip/hip_runtime.h>

typedef __attribute__((ext_vector_type(16))) _Float16 v16h;
typedef __attribute__((ext_vector_type(8)))  float    v8f;
typedef __attribute__((ext_vector_type(4)))  float    f32x4;

#define BATCH      65536
#define TREAT      64
#define INDIM      256
#define OUTDIM     256
#define NB         5
#define ROW_STRIDE 320   // TREAT + INDIM

// ---------------------------------------------------------------------------
// Kernel 1: per-row basis computation + treat-block passthrough copy.
// ---------------------------------------------------------------------------
__global__ void prep_basis(const float* __restrict__ x,
                           float* __restrict__ out,
                           float* __restrict__ basisWS) {
  const int row = blockIdx.x * blockDim.x + threadIdx.x;
  const float* xr  = x   + (size_t)row * ROW_STRIDE;
  float*      orow = out + (size_t)row * ROW_STRIDE;
  float sum = 0.0f;
#pragma unroll
  for (int i = 0; i < 16; ++i) {
    f32x4 v = *(const f32x4*)(xr + i * 4);
    *(f32x4*)(orow + i * 4) = v;          // passthrough of x_treat
    sum += (v.x + v.y) + (v.z + v.w);
  }
  const float t  = sum * (1.0f / 64.0f);
  const float r1 = fmaxf(t - 0.33f, 0.0f);
  const float r2 = fmaxf(t - 0.66f, 0.0f);
  float* b = basisWS + (size_t)row * NB;
  b[0] = 1.0f;
  b[1] = t;
  b[2] = t * t;
  b[3] = r1 * r1;
  b[4] = r2 * r2;
}

// ---------------------------------------------------------------------------
// Kernel 2: pack weight (IND, OUTD, NB) f32 -> f16 WMMA B-fragments.
// Fragment f = ((d*16 + ntile)*8 + ktile): 32x16 (KxN) tile of W_d in exact
// v_wmma_f32_16x16x32_f16 B layout; each lane's 16 halfs contiguous (32B).
//   lane L (g=L/16, n=L%16): half v -> K = ktile*32 + g*8 + (v&7) + (v>>3)*16, N = ntile*16+n
// ---------------------------------------------------------------------------
__global__ void pack_weight(const float* __restrict__ w,
                            _Float16* __restrict__ bw) {
  const int tid  = blockIdx.x * blockDim.x + threadIdx.x;  // 640 frags * 32
  const int lane = tid & 31;
  const int frag = tid >> 5;
  const int ktile = frag & 7;
  const int rest  = frag >> 3;
  const int ntile = rest & 15;
  const int d     = rest >> 4;
  const int g     = lane >> 4;
  const int o     = ntile * 16 + (lane & 15);
  const int kbase = ktile * 32 + g * 8;
  v16h v;
#pragma unroll
  for (int e = 0; e < 16; ++e) {
    const int i = kbase + (e & 7) + (e >> 3) * 16;  // K index == input-feature index
    v[e] = (_Float16)w[((size_t)i * OUTDIM + o) * NB + d];
  }
  *(v16h*)(bw + (size_t)frag * 512 + (size_t)lane * 16) = v;
}

// ---------------------------------------------------------------------------
// Kernel 3: WMMA GEMM with 5 accumulators per output tile.
//   acc_d = X_f16 @ W_d    (8 ktiles, f32 accumulate)
//   out[b,o] = relu( sum_d basis[b,d] * (acc_d[b,o] + bias[o,d]) )
// Block = 256 threads = 8 waves covering 32 rows x 128 cols.
// Wave: 16 rows x 32 cols = 2 n-tiles x 5 basis accumulators (10x v8f).
// All B addresses are loop-invariant-base + compile-time immediate offsets
// (d*131072 + kt*1024 bytes), so loads hoist freely with no address VALU.
// 80 v_wmma per wave; no basis pk_mul on the critical path.
// ---------------------------------------------------------------------------
__global__ __launch_bounds__(256)
void gemm_dynfc(const float* __restrict__ x,
                const float* __restrict__ bias,
                const float* __restrict__ basisWS,
                const _Float16* __restrict__ bw,
                float* __restrict__ out) {
  const int lane = threadIdx.x & 31;
  const int wave = threadIdx.x >> 5;
  const int g    = lane >> 4;     // half-wave group
  const int ln   = lane & 15;
  const int wm   = wave & 1;      // 2 m-waves
  const int wn   = wave >> 1;     // 4 n-waves

  const int row0 = blockIdx.x * 32 + wm * 16;
  const int nt0  = blockIdx.y * 8 + wn * 2;   // global 16-col n-tile index

  // Loop-invariant per-lane bases; all per-iteration offsets are constants.
  const float*    abase = x  + (size_t)(row0 + ln) * ROW_STRIDE + TREAT + g * 8;
  const _Float16* bb0   = bw + (size_t)nt0 * 4096 + (size_t)lane * 16;
  const _Float16* bb1   = bb0 + 4096;        // nt0+1

  v8f acc0[NB];
  v8f acc1[NB];
#pragma unroll
  for (int d = 0; d < NB; ++d) { acc0[d] = (v8f){}; acc1[d] = (v8f){}; }

#pragma unroll
  for (int kt = 0; kt < 8; ++kt) {
    // A layout (16-bit, 16x32): lane (g, m=ln) holds K = kt*32+g*8+[0..7] and
    // K = kt*32+16+g*8+[0..7] -> two contiguous 8-float runs, const offsets.
    const float* ap = abase + kt * 32;
    f32x4 f0 = *(const f32x4*)(ap);
    f32x4 f1 = *(const f32x4*)(ap + 4);
    f32x4 f2 = *(const f32x4*)(ap + 16);
    f32x4 f3 = *(const f32x4*)(ap + 20);
    v16h a;
    a[0]  = (_Float16)f0.x; a[1]  = (_Float16)f0.y;
    a[2]  = (_Float16)f0.z; a[3]  = (_Float16)f0.w;
    a[4]  = (_Float16)f1.x; a[5]  = (_Float16)f1.y;
    a[6]  = (_Float16)f1.z; a[7]  = (_Float16)f1.w;
    a[8]  = (_Float16)f2.x; a[9]  = (_Float16)f2.y;
    a[10] = (_Float16)f2.z; a[11] = (_Float16)f2.w;
    a[12] = (_Float16)f3.x; a[13] = (_Float16)f3.y;
    a[14] = (_Float16)f3.z; a[15] = (_Float16)f3.w;

#pragma unroll
    for (int d = 0; d < NB; ++d) {
      const int off = d * 65536 + kt * 512;  // halfs; byte offset fits imm24
      const v16h b0 = *(const v16h*)(bb0 + off);
      const v16h b1 = *(const v16h*)(bb1 + off);
      acc0[d] = __builtin_amdgcn_wmma_f32_16x16x32_f16(
          false, a, false, b0, (short)0, acc0[d], false, false);
      acc1[d] = __builtin_amdgcn_wmma_f32_16x16x32_f16(
          false, a, false, b1, (short)0, acc1[d], false, false);
    }
  }

  // Epilogue: out = relu( sum_d basis[row,d] * (acc_d + bias[o,d]) ),
  // stored at concat offset col = 64 + o.
  const int o0 = nt0 * 16 + ln;
  const int o1 = o0 + 16;
  float bias0[NB], bias1[NB];
#pragma unroll
  for (int d = 0; d < NB; ++d) {
    bias0[d] = bias[(size_t)o0 * NB + d];
    bias1[d] = bias[(size_t)o1 * NB + d];
  }

#pragma unroll
  for (int r = 0; r < 8; ++r) {
    // C/D layout: VGPR r, lanes 0-15 -> M=r, lanes 16-31 -> M=r+8; N=lane%16
    const int row = row0 + r + 8 * g;
    const float* br = basisWS + (size_t)row * NB;
    float t0 = 0.0f;
    float t1 = 0.0f;
#pragma unroll
    for (int d = 0; d < NB; ++d) {
      const float bb = br[d];
      t0 += bb * (acc0[d][r] + bias0[d]);
      t1 += bb * (acc1[d][r] + bias1[d]);
    }
    t0 = fmaxf(t0, 0.0f);
    t1 = fmaxf(t1, 0.0f);
    float* orow = out + (size_t)row * ROW_STRIDE + TREAT;
    orow[o0] = t0;
    orow[o1] = t1;
  }
}

// ---------------------------------------------------------------------------
extern "C" void kernel_launch(void* const* d_in, const int* in_sizes, int n_in,
                              void* d_out, int out_size, void* d_ws, size_t ws_size,
                              hipStream_t stream) {
  const float* x      = (const float*)d_in[0];   // (65536, 320) f32
  const float* weight = (const float*)d_in[1];   // (256, 256, 5) f32
  const float* bias   = (const float*)d_in[2];   // (256, 5) f32
  float* out = (float*)d_out;                    // (65536, 320) f32

  // Workspace: [ basis (B x 5 f32) | packed weight (640 frags x 1KB f16) ]
  float*    basisWS = (float*)d_ws;
  _Float16* bw      = (_Float16*)((char*)d_ws + (size_t)BATCH * NB * sizeof(float));

  prep_basis<<<BATCH / 256, 256, 0, stream>>>(x, out, basisWS);
  pack_weight<<<(640 * 32) / 256, 256, 0, stream>>>(weight, bw);
  gemm_dynfc<<<dim3(BATCH / 32, 2), 256, 0, stream>>>(x, bias, basisWS, bw, out);
}
